// SaaDecoder_45140106280984
// MI455X (gfx1250) — matrix operand order repaired and checked by reading, not compiled, not run on hardware
//
#include <hip/hip_runtime.h>
#include <hip/hip_bf16.h>

typedef __attribute__((ext_vector_type(16))) _Float16 v16h;
typedef __attribute__((ext_vector_type(2)))  _Float16 h2v;
typedef __attribute__((ext_vector_type(8)))  float    v8f;

#define BATCH 32
#define HID   512
#define VOCAB 32000
#define TDEC  64
#define TENC  512

// ---------------------------------------------------------------------------
// Weight conversion: fp32 -> f16, with optional horizontal concat of two
// row-major matrices (W_ih | W_hh) so the GEMM sees one contiguous K.
// ---------------------------------------------------------------------------
__global__ void cvt_cat_f16(_Float16* __restrict__ dst,
                            const float* __restrict__ A, int K1,
                            const float* __restrict__ Bm, int K2, int N) {
  long idx = (long)blockIdx.x * blockDim.x + threadIdx.x;
  long K = K1 + K2;
  if (idx >= (long)N * K) return;
  long n = idx / K;
  int  k = (int)(idx - n * K);
  float v = (k < K1) ? A[n * K1 + k] : Bm[n * K2 + (k - K1)];
  dst[idx] = (_Float16)v;
}

__global__ void cvt_f16(_Float16* __restrict__ dst, const float* __restrict__ src, long n) {
  long idx = (long)blockIdx.x * blockDim.x + threadIdx.x;
  if (idx < n) dst[idx] = (_Float16)src[idx];
}

// ---------------------------------------------------------------------------
// State init: zero h0,c0,h1,c1,ctx (contiguous) and seed cur tokens.
// ---------------------------------------------------------------------------
__global__ void init_state(float* __restrict__ states, int* __restrict__ cur,
                           const int* __restrict__ dec_in) {
  int idx = blockIdx.x * blockDim.x + threadIdx.x;
  if (idx < 5 * BATCH * HID) states[idx] = 0.0f;
  if (idx < BATCH) cur[idx] = dec_in[idx * TDEC];  // decoder_input[:,0]
}

// ---------------------------------------------------------------------------
// Build LSTM0 input [emb(cur) | ctx | h0] as f16 (32 x 1536).
// ---------------------------------------------------------------------------
__global__ void prep_x0(_Float16* __restrict__ x16,
                        const float* __restrict__ emb, const int* __restrict__ cur,
                        const float* __restrict__ ctx, const float* __restrict__ h0) {
  int idx = blockIdx.x * blockDim.x + threadIdx.x;  // < 32*1536
  if (idx >= BATCH * 1536) return;
  int m = idx / 1536, c = idx - m * 1536;
  float v;
  if (c < HID)            v = emb[(long)cur[m] * HID + c];
  else if (c < 2 * HID)   v = ctx[m * HID + (c - HID)];
  else                    v = h0 [m * HID + (c - 2 * HID)];
  x16[idx] = (_Float16)v;
}

// ---------------------------------------------------------------------------
// WMMA GEMM:  C(32 x N) = X16(32 x K) @ W(N x K)^T  [+ bias]
// One wave per 16-column tile, full M=32 via two accumulators.
// NT=true streams C past L2 (used for the 262 MB logits output so the
// L2-resident weight working set is not evicted).
// ---------------------------------------------------------------------------
__device__ inline v16h load_a_frag(const _Float16* __restrict__ X, int K,
                                   int m0, int k, int lane) {
  // 16-bit A 16x32 layout: lane<16 -> K {0..7,16..23}; lane>=16 -> K {8..15,24..31}
  int row  = m0 + (lane & 15);
  int koff = (lane & 16) ? 8 : 0;
  const _Float16* p = X + (long)row * K + k + koff;
  union { v16h v; uint4 q[2]; } u;
  u.q[0] = *(const uint4*)(p);
  u.q[1] = *(const uint4*)(p + 16);
  return u.v;
}

__device__ inline v16h load_b_frag(const _Float16* __restrict__ W, int K,
                                   int n0, int k, int lane) {
  // 16-bit B 32x16 layout: lane = column; lanes 0-15 hold K {0..7,16..23}, 16-31 hold K {8..15,24..31}
  int col  = n0 + (lane & 15);     // B column == weight row (x @ W^T)
  int koff = (lane & 16) ? 8 : 0;
  const _Float16* p = W + (long)col * K + k + koff;
  union { v16h v; uint4 q[2]; } u;
  u.q[0] = *(const uint4*)(p);
  u.q[1] = *(const uint4*)(p + 16);
  return u.v;
}

template <bool NT>
__global__ void gemm_wmma(const _Float16* __restrict__ X,
                          const _Float16* __restrict__ W,
                          float* __restrict__ C,
                          const float* __restrict__ bias,
                          int N, int K, long ldC) {
  int wave = threadIdx.x >> 5;
  int lane = threadIdx.x & 31;
  int n0 = (blockIdx.x * 4 + wave) * 16;
  if (n0 >= N) return;  // wave-uniform; WMMA waves keep EXEC all-ones

  v8f acc0 = {}; v8f acc1 = {};
  for (int k = 0; k < K; k += 32) {
    v16h a0 = load_a_frag(X, K, 0,  k, lane);
    v16h a1 = load_a_frag(X, K, 16, k, lane);
    v16h b  = load_b_frag(W, K, n0, k, lane);
    acc0 = __builtin_amdgcn_wmma_f32_16x16x32_f16(false, a0, false, b,
                                                  (short)0, acc0, false, false);
    acc1 = __builtin_amdgcn_wmma_f32_16x16x32_f16(false, a1, false, b,
                                                  (short)0, acc1, false, false);
  }
  // C/D layout: VGPR r -> M = (lane<16 ? r : r+8); N = lane&15
  int n = n0 + (lane & 15);
  int mb = (lane & 16) ? 8 : 0;
  float bv = bias ? bias[n] : 0.0f;
#pragma unroll
  for (int r = 0; r < 8; ++r) {
    float v0 = acc0[r] + bv;
    float v1 = acc1[r] + bv;
    float* p0 = C + (long)(mb + r)      * ldC + n;
    float* p1 = C + (long)(mb + r + 16) * ldC + n;
    if (NT) { __builtin_nontemporal_store(v0, p0); __builtin_nontemporal_store(v1, p1); }
    else    { *p0 = v0; *p1 = v1; }
  }
}

// ---------------------------------------------------------------------------
// LSTM pointwise: gates (32x2048, i|f|g|o) -> update c,h. Optionally stage the
// next GEMM input x16 = [h_new | h_other] (f16, 32x1024).
// ---------------------------------------------------------------------------
__device__ inline float sigm(float x) { return 1.0f / (1.0f + expf(-x)); }

__global__ void lstm_act(const float* __restrict__ gates,
                         const float* __restrict__ b_ih, const float* __restrict__ b_hh,
                         float* __restrict__ c, float* __restrict__ h,
                         _Float16* __restrict__ xnext, const float* __restrict__ hother) {
  int idx = blockIdx.x * blockDim.x + threadIdx.x;  // < 32*512
  if (idx >= BATCH * HID) return;
  int m = idx >> 9, j = idx & 511;
  const float* g = gates + (long)m * 2048;
  float gi = g[j]        + b_ih[j]        + b_hh[j];
  float gf = g[512 + j]  + b_ih[512 + j]  + b_hh[512 + j];
  float gg = g[1024 + j] + b_ih[1024 + j] + b_hh[1024 + j];
  float go = g[1536 + j] + b_ih[1536 + j] + b_hh[1536 + j];
  float cn = sigm(gf) * c[idx] + sigm(gi) * tanhf(gg);
  float hn = sigm(go) * tanhf(cn);
  c[idx] = cn;
  h[idx] = hn;
  if (xnext) {
    xnext[(long)m * 1024 + j]       = (_Float16)hn;
    xnext[(long)m * 1024 + 512 + j] = (_Float16)hother[idx];
  }
}

// ---------------------------------------------------------------------------
// Attention, split for parallelism. enc16 is the f16 copy of encoder_outputs.
// ---------------------------------------------------------------------------
// scores: grid = BATCH*8 blocks, 256 threads; wave per encoder row.
__global__ void attn_scores(const float* __restrict__ h1,
                            const _Float16* __restrict__ enc16,
                            const unsigned char* __restrict__ mask,
                            float* __restrict__ sbuf) {
  __shared__ float sh[HID];
  int b   = blockIdx.x >> 3;
  int blk = blockIdx.x & 7;           // covers t in [blk*64, blk*64+64)
  for (int i = threadIdx.x; i < HID; i += 256) sh[i] = h1[b * HID + i];
  __syncthreads();
  int wave = threadIdx.x >> 5, lane = threadIdx.x & 31;
  const float scale = 0.044194173824159216f;  // 1/sqrt(512)
  for (int tt = wave; tt < 64; tt += 8) {
    int t = blk * 64 + tt;
    const h2v* e2 = (const h2v*)(enc16 + ((long)b * TENC + t) * HID);
    float s = 0.0f;
#pragma unroll
    for (int j = 0; j < 8; ++j) {
      int p = lane + 32 * j;          // pair index, 256 pairs per row
      h2v pr = e2[p];
      s += (float)pr.x * sh[2 * p] + (float)pr.y * sh[2 * p + 1];
    }
    for (int off = 16; off > 0; off >>= 1) s += __shfl_down(s, off, 32);
    if (lane == 0) {
      s *= scale;
      if (!mask[(long)b * TENC + t]) s = -3.402823466e38f;
      sbuf[b * TENC + t] = s;
    }
  }
}

// softmax in place on sbuf rows: grid = BATCH blocks, 256 threads.
__global__ void attn_softmax(float* __restrict__ sbuf) {
  __shared__ float red[256];
  int b = blockIdx.x, tid = threadIdx.x;
  float* s = sbuf + b * TENC;
  float a0 = s[tid], a1 = s[tid + 256];
  red[tid] = fmaxf(a0, a1);
  __syncthreads();
  for (int st = 128; st > 0; st >>= 1) {
    if (tid < st) red[tid] = fmaxf(red[tid], red[tid + st]);
    __syncthreads();
  }
  float mx = red[0];
  __syncthreads();
  float e0 = expf(a0 - mx), e1 = expf(a1 - mx);
  red[tid] = e0 + e1;
  __syncthreads();
  for (int st = 128; st > 0; st >>= 1) {
    if (tid < st) red[tid] += red[tid + st];
    __syncthreads();
  }
  float inv = 1.0f / red[0];
  s[tid] = e0 * inv;
  s[tid + 256] = e1 * inv;
}

// context: grid = BATCH*2 blocks, 256 threads; thread per hidden column.
// Also stages MLP1 input x16 = [h1 | ctx] (f16, 32x1024).
__global__ void attn_ctx(const float* __restrict__ sbuf,
                         const _Float16* __restrict__ enc16,
                         const float* __restrict__ h1,
                         float* __restrict__ ctx, _Float16* __restrict__ x16) {
  __shared__ float w[TENC];
  int b  = blockIdx.x >> 1;
  int hc = blockIdx.x & 1;
  int tid = threadIdx.x;
  w[tid] = sbuf[b * TENC + tid];
  w[tid + 256] = sbuf[b * TENC + tid + 256];
  __syncthreads();
  int hcol = hc * 256 + tid;
  const _Float16* e = enc16 + (long)b * TENC * HID + hcol;
  float acc = 0.0f;
  for (int t = 0; t < TENC; ++t)
    acc += w[t] * (float)e[(long)t * HID];
  ctx[b * HID + hcol] = acc;
  x16[(long)b * 1024 + hcol]       = (_Float16)h1[b * HID + hcol];
  x16[(long)b * 1024 + 512 + hcol] = (_Float16)acc;
}

// ---------------------------------------------------------------------------
// MLP hidden: tanh(lin + b1) staged as f16 (32x512) for the logits GEMM.
// ---------------------------------------------------------------------------
__global__ void mlp_act(const float* __restrict__ lin, const float* __restrict__ b1,
                        _Float16* __restrict__ x16) {
  int idx = blockIdx.x * blockDim.x + threadIdx.x;  // < 32*512
  if (idx >= BATCH * HID) return;
  int j = idx & 511;
  x16[idx] = (_Float16)tanhf(lin[idx] + b1[j]);
}

// ---------------------------------------------------------------------------
// argmax over V=32000 logits per batch row -> next token.
// ---------------------------------------------------------------------------
__global__ void argmax_kernel(const float* __restrict__ logits, long ldRow,
                              int* __restrict__ cur) {
  __shared__ float bv[256];
  __shared__ int   bidx[256];
  int b = blockIdx.x, tid = threadIdx.x;
  const float* p = logits + (long)b * ldRow;
  float best = -3.402823466e38f; int bi = 0x7fffffff;
  for (int i = tid; i < VOCAB; i += 256) {
    float v = p[i];
    if (v > best) { best = v; bi = i; }
  }
  bv[tid] = best; bidx[tid] = bi;
  __syncthreads();
  for (int s = 128; s > 0; s >>= 1) {
    if (tid < s) {
      if (bv[tid + s] > bv[tid] ||
          (bv[tid + s] == bv[tid] && bidx[tid + s] < bidx[tid])) {
        bv[tid] = bv[tid + s]; bidx[tid] = bidx[tid + s];
      }
    }
    __syncthreads();
  }
  if (tid == 0) cur[b] = bidx[0];
}

// ---------------------------------------------------------------------------
// Launch: convert weights + encoder once, then 64 sequential decode steps.
// ---------------------------------------------------------------------------
extern "C" void kernel_launch(void* const* d_in, const int* in_sizes, int n_in,
                              void* d_out, int out_size, void* d_ws, size_t ws_size,
                              hipStream_t stream) {
  const int*   dec_in  = (const int*)  d_in[0];
  const float* enc     = (const float*)d_in[1];
  const unsigned char* mask = (const unsigned char*)d_in[2];
  const float* emb     = (const float*)d_in[3];
  const float* W_ih0   = (const float*)d_in[4];
  const float* W_hh0   = (const float*)d_in[5];
  const float* b_ih0   = (const float*)d_in[6];
  const float* b_hh0   = (const float*)d_in[7];
  const float* W_ih1   = (const float*)d_in[8];
  const float* W_hh1   = (const float*)d_in[9];
  const float* b_ih1   = (const float*)d_in[10];
  const float* b_hh1   = (const float*)d_in[11];
  const float* mlp_W1  = (const float*)d_in[12];
  const float* mlp_b1  = (const float*)d_in[13];
  const float* mlp_W2  = (const float*)d_in[14];
  const float* mlp_b2  = (const float*)d_in[15];
  float* out = (float*)d_out;

  // bump allocator over d_ws
  char* ws = (char*)d_ws;
  auto alloc = [&](size_t bytes) -> void* {
    void* p = (void*)ws;
    ws += (bytes + 255) & ~(size_t)255;
    return p;
  };
  _Float16* x16     = (_Float16*)alloc(BATCH * 1536 * sizeof(_Float16));
  float*    gates0  = (float*)   alloc(BATCH * 2048 * sizeof(float));
  float*    gates1  = (float*)   alloc(BATCH * 2048 * sizeof(float));
  float*    mlp_lin = (float*)   alloc(BATCH * HID  * sizeof(float));
  float*    sbuf    = (float*)   alloc(BATCH * TENC * sizeof(float));
  float*    states  = (float*)   alloc(5 * BATCH * HID * sizeof(float));
  int*      cur     = (int*)     alloc(BATCH * sizeof(int));
  _Float16* Wcat0   = (_Float16*)alloc((size_t)2048 * 1536 * sizeof(_Float16));
  _Float16* Wcat1   = (_Float16*)alloc((size_t)2048 * 1024 * sizeof(_Float16));
  _Float16* W1f     = (_Float16*)alloc((size_t)HID * 1024 * sizeof(_Float16));
  _Float16* W2f     = (_Float16*)alloc((size_t)VOCAB * HID * sizeof(_Float16));
  _Float16* enc16   = (_Float16*)alloc((size_t)BATCH * TENC * HID * sizeof(_Float16));

  float* h0  = states + 0 * BATCH * HID;
  float* c0  = states + 1 * BATCH * HID;
  float* h1  = states + 2 * BATCH * HID;
  float* c1  = states + 3 * BATCH * HID;
  float* ctx = states + 4 * BATCH * HID;

  // ---- one-time conversions (inside the graph; amortized over 64 steps)
  {
    long n;
    n = (long)2048 * 1536;
    cvt_cat_f16<<<(int)((n + 255) / 256), 256, 0, stream>>>(Wcat0, W_ih0, 1024, W_hh0, 512, 2048);
    n = (long)2048 * 1024;
    cvt_cat_f16<<<(int)((n + 255) / 256), 256, 0, stream>>>(Wcat1, W_ih1, 512, W_hh1, 512, 2048);
    n = (long)HID * 1024;
    cvt_f16<<<(int)((n + 255) / 256), 256, 0, stream>>>(W1f, mlp_W1, n);
    n = (long)VOCAB * HID;
    cvt_f16<<<(int)((n + 255) / 256), 256, 0, stream>>>(W2f, mlp_W2, n);
    n = (long)BATCH * TENC * HID;
    cvt_f16<<<(int)((n + 255) / 256), 256, 0, stream>>>(enc16, enc, n);
  }
  init_state<<<(5 * BATCH * HID + 255) / 256, 256, 0, stream>>>(states, cur, dec_in);

  const long ldOut = (long)TDEC * VOCAB;
  for (int t = 0; t < TDEC; ++t) {
    // x = [emb(cur) | ctx | h0]
    prep_x0<<<(BATCH * 1536 + 255) / 256, 256, 0, stream>>>(x16, emb, cur, ctx, h0);
    // LSTM0 gates: (32x1536) @ (2048x1536)^T
    gemm_wmma<false><<<2048 / 64, 128, 0, stream>>>(x16, Wcat0, gates0, nullptr, 2048, 1536, 2048);
    // c0,h0 update; stage [h0_new | h1_old] for LSTM1
    lstm_act<<<(BATCH * HID + 255) / 256, 256, 0, stream>>>(gates0, b_ih0, b_hh0, c0, h0, x16, h1);
    // LSTM1 gates: (32x1024) @ (2048x1024)^T
    gemm_wmma<false><<<2048 / 64, 128, 0, stream>>>(x16, Wcat1, gates1, nullptr, 2048, 1024, 2048);
    lstm_act<<<(BATCH * HID + 255) / 256, 256, 0, stream>>>(gates1, b_ih1, b_hh1, c1, h1, nullptr, nullptr);
    // attention: scores -> softmax -> context (stages [h1 | ctx])
    attn_scores<<<BATCH * 8, 256, 0, stream>>>(h1, enc16, mask, sbuf);
    attn_softmax<<<BATCH, 256, 0, stream>>>(sbuf);
    attn_ctx<<<BATCH * 2, 256, 0, stream>>>(sbuf, enc16, h1, ctx, x16);
    // MLP hidden: (32x1024) @ (512x1024)^T
    gemm_wmma<false><<<HID / 64, 128, 0, stream>>>(x16, W1f, mlp_lin, nullptr, HID, 1024, HID);
    mlp_act<<<(BATCH * HID + 255) / 256, 256, 0, stream>>>(mlp_lin, mlp_b1, x16);
    // logits: (32x512) @ (32000x512)^T + b2, NT-streamed into d_out[:, t, :]
    gemm_wmma<true><<<VOCAB / 64, 128, 0, stream>>>(x16, W2f, out + (long)t * VOCAB, mlp_b2,
                                                    VOCAB, HID, ldOut);
    // next token
    argmax_kernel<<<BATCH, 256, 0, stream>>>(out + (long)t * VOCAB, ldOut, cur);
  }
}
